// Block_56530359550411
// MI455X (gfx1250) — matrix operand order repaired and checked by reading, not compiled, not run on hardware
//
#include <hip/hip_runtime.h>
#include <stdint.h>

// ---------------- types ----------------
typedef __attribute__((ext_vector_type(16))) __bf16 v16bf;
typedef __attribute__((ext_vector_type(8)))  float  v8f;

struct alignas(16) U4 { unsigned x, y, z, w; };
struct alignas(8)  U2 { unsigned x, y; };
union FragBF { v16bf v; U4 q[2]; };            // 16 bf16 = 32B (A or B operand)
union FragF  { v8f  v; float f[8]; };          // 16x16 f32 C/D tile (8 VGPRs)

#define T_SEQ 2048
#define DIM   2048
#define NH    16
#define HD    128
#define FFN   8192
#define E3    6144
#define RMS_EPS 1.1920929e-07f

__device__ __forceinline__ unsigned short f2bf(float f) {
  union { float f; unsigned u; } v; v.f = f;
  unsigned u = v.u;
  unsigned r = u + 0x7FFFu + ((u >> 16) & 1u);   // round-nearest-even
  return (unsigned short)(r >> 16);
}
__device__ __forceinline__ float bf2f(unsigned short b) {
  union { unsigned u; float f; } v; v.u = ((unsigned)b) << 16;
  return v.f;
}

// ---------------- weight prep ----------------
__global__ void convert_scale_kernel(const float* __restrict__ src,
                                     unsigned short* __restrict__ dst, int n,
                                     const float* __restrict__ lam, int lidx) {
  int i = blockIdx.x * 256 + threadIdx.x;
  if (i < n) {
    float s = lam ? lam[lidx] : 1.0f;
    dst[i] = f2bf(src[i] * s);
  }
}

// c_proj (FFN, DIM) -> (DIM, FFN) bf16 via 32x32 LDS tiles (coalesced both sides)
__global__ __launch_bounds__(256) void transpose_convert_kernel(const float* __restrict__ src,
                                                                unsigned short* __restrict__ dst) {
  __shared__ float tile[32][33];
  int f0 = blockIdx.x * 32;       // source row tile (FFN dim)
  int d0 = blockIdx.y * 32;       // source col tile (DIM dim)
  int r = threadIdx.x >> 3, c4 = (threadIdx.x & 7) * 4;
  const float* sp = src + (size_t)(f0 + r) * DIM + d0 + c4;
  for (int j = 0; j < 4; j++) tile[r][c4 + j] = sp[j];
  __syncthreads();
  union { U2 q; unsigned short s[4]; } pk;
  for (int j = 0; j < 4; j++) pk.s[j] = f2bf(tile[c4 + j][r]);   // dst[d0+r][f0+c4+j]
  *(U2*)&dst[(size_t)(d0 + r) * FFN + f0 + c4] = pk.q;
}

// ---------------- rmsnorm (row of 2048) -> bf16 ----------------
__global__ __launch_bounds__(256) void rmsnorm_kernel(const float* __restrict__ x,
                                                      unsigned short* __restrict__ out) {
  int t = blockIdx.x;
  const float* xp = x + (size_t)t * DIM;
  float vals[8], s = 0.f;
  for (int i = 0; i < 8; i++) { vals[i] = xp[threadIdx.x + i * 256]; s += vals[i] * vals[i]; }
  for (int m = 16; m >= 1; m >>= 1) s += __shfl_xor(s, m, 32);
  __shared__ float red[8];
  int lane = threadIdx.x & 31, w = threadIdx.x >> 5;
  if (lane == 0) red[w] = s;
  __syncthreads();
  float tot = 0.f;
  for (int i = 0; i < 8; i++) tot += red[i];
  float sc = rsqrtf(tot * (1.0f / DIM) + RMS_EPS);
  for (int i = 0; i < 8; i++)
    out[(size_t)t * DIM + threadIdx.x + i * 256] = f2bf(vals[i] * sc);
}

// ---------------- attention / ve gates: sigmoid(xn . g_w[h]) ----------------
__global__ __launch_bounds__(256) void gates_kernel(const unsigned short* __restrict__ xn,
                                                    const float* __restrict__ agw,
                                                    const float* __restrict__ vgw,
                                                    float* __restrict__ agate,
                                                    float* __restrict__ vgate) {
  int t = blockIdx.x;
  int h = threadIdx.x >> 4, part = threadIdx.x & 15;
  float sa = 0.f, sv = 0.f;
  const unsigned short* xp = xn + (size_t)t * DIM;
  for (int i = 0; i < 128; i++) {
    int d = i * 16 + part;                 // consecutive lanes -> consecutive addresses
    float xv = bf2f(xp[d]);
    sa += xv * agw[h * DIM + d];
    sv += xv * vgw[h * DIM + d];
  }
  for (int m = 1; m < 16; m <<= 1) { sa += __shfl_xor(sa, m, 32); sv += __shfl_xor(sv, m, 32); }
  if (part == 0) {
    agate[t * NH + h] = 1.f / (1.f + __expf(-sa));
    vgate[t * NH + h] = 1.f / (1.f + __expf(-sv));   // VE_GATE_SCALE = 1
  }
}

// ---------------- q/k rmsnorm + rotary + key-shift ---------------------------------
__global__ __launch_bounds__(128) void qk_prep(const float* __restrict__ qkv,
                                               const float* __restrict__ cosb,
                                               const float* __restrict__ sinb,
                                               const int* __restrict__ key_offset,
                                               unsigned short* __restrict__ qo,
                                               unsigned short* __restrict__ ko) {
  const int t = blockIdx.x, h = blockIdx.y, d = threadIdx.x;
  const int tp = (t > 0) ? t - 1 : 0;
  __shared__ float qn[HD], kn[HD], kpn[HD];
  __shared__ float red[12];
  float qv  = qkv[(size_t)t * E3 + h * HD + d];
  float kv  = qkv[(size_t)t * E3 + DIM + h * HD + d];
  float kpv = qkv[(size_t)tp * E3 + DIM + h * HD + d];
  float s0 = qv * qv, s1 = kv * kv, s2 = kpv * kpv;
  int lane = threadIdx.x & 31, w = threadIdx.x >> 5;
  for (int m = 16; m >= 1; m >>= 1) {
    s0 += __shfl_xor(s0, m, 32); s1 += __shfl_xor(s1, m, 32); s2 += __shfl_xor(s2, m, 32);
  }
  if (lane == 0) { red[w] = s0; red[4 + w] = s1; red[8 + w] = s2; }
  __syncthreads();
  float t0 = red[0] + red[1] + red[2] + red[3];
  float t1 = red[4] + red[5] + red[6] + red[7];
  float t2 = red[8] + red[9] + red[10] + red[11];
  float rq  = rsqrtf(t0 * (1.f / HD) + RMS_EPS);
  float rk  = rsqrtf(t1 * (1.f / HD) + RMS_EPS);
  float rkp = rsqrtf(t2 * (1.f / HD) + RMS_EPS);
  qn[d] = qv * rq; kn[d] = kv * rk; kpn[d] = kpv * rkp;
  __syncthreads();
  float qr, kr, kpr;
  if (d < 64) {
    float c = cosb[t * 64 + d], s = sinb[t * 64 + d];
    qr = qn[d] * c + qn[d + 64] * s;
    kr = kn[d] * c + kn[d + 64] * s;
    float cp = cosb[tp * 64 + d], sp = sinb[tp * 64 + d];
    kpr = kpn[d] * cp + kpn[d + 64] * sp;
  } else {
    int dd = d - 64;
    float c = cosb[t * 64 + dd], s = sinb[t * 64 + dd];
    qr = -qn[dd] * s + qn[d] * c;
    kr = -kn[dd] * s + kn[d] * c;
    float cp = cosb[tp * 64 + dd], sp = sinb[tp * 64 + dd];
    kpr = -kpn[dd] * sp + kpn[d] * cp;
  }
  float kf = kr;
  if (key_offset[0] != 0) {
    int quarter = d >> 5;                 // quarters 1,3 come from the shifted row
    if (quarter == 1 || quarter == 3) kf = kpr;
  }
  qo[((size_t)t * NH + h) * HD + d] = f2bf(qr);
  ko[((size_t)t * NH + h) * HD + d] = f2bf(kf);
}

// ---------------- v = qkv_v + gate*ve, written transposed (H,HD,S) via LDS tiles ---
__global__ __launch_bounds__(256) void v_prep(const float* __restrict__ qkv,
                                              const float* __restrict__ ve,
                                              const float* __restrict__ vgate,
                                              unsigned short* __restrict__ vto) {
  __shared__ unsigned short tile[64][136];
  const int t0 = blockIdx.x * 64, h = blockIdx.y;
  int d = threadIdx.x & 127, tt2 = threadIdx.x >> 7;   // 2 t-rows per pass
  for (int p = 0; p < 32; p++) {
    int tl = p * 2 + tt2, t = t0 + tl;
    float vv = qkv[(size_t)t * E3 + 2 * DIM + h * HD + d];
    float vf = vv + vgate[t * NH + h] * ve[(size_t)t * DIM + h * HD + d];
    tile[tl][d] = f2bf(vf);
  }
  __syncthreads();
  int dRow = threadIdx.x >> 1, half = threadIdx.x & 1;
  union { U4 q; unsigned short s[8]; } pk;
  for (int j = 0; j < 4; j++) {
    for (int e = 0; e < 8; e++) pk.s[e] = tile[half * 32 + j * 8 + e][dRow];
    *(U4*)&vto[(size_t)(h * HD + dRow) * T_SEQ + t0 + half * 32 + j * 8] = pk.q;
  }
}

// ---------------- generic bf16 WMMA GEMM: C(MxN) = A(MxK) · B(NxK)^T --------------
// Ping-pong LDS double buffering; global->reg prefetch overlaps the WMMA block.
// EPI: 0 = f32 out, 1 = f32 out + residual, 2 = relu()^2 -> bf16 out
template <int EPI>
__global__ __launch_bounds__(256) void gemm_nt(const unsigned short* __restrict__ A, int lda,
                                               const unsigned short* __restrict__ B, int ldb,
                                               float* __restrict__ Cf,
                                               unsigned short* __restrict__ Cbf, int ldc,
                                               const float* __restrict__ resid, int K) {
  __shared__ alignas(16) unsigned short lA[2][128 * 40];  // 80B row stride: conflict-free b128
  __shared__ alignas(16) unsigned short lB[2][128 * 40];
  const int tid = threadIdx.x;
  const int wid = tid >> 5, lane = tid & 31;
  const int laneN = lane & 15, laneH = lane >> 4;
  const int mBase = blockIdx.y * 128, nBase = blockIdx.x * 128;
  const int wm = (wid & 3) * 32, wn = (wid >> 2) * 64;
  const int r0 = tid >> 2, c0 = (tid & 3) * 8;            // staging: rows r0, r0+64

  const unsigned short* Ap = A + (size_t)mBase * lda;
  const unsigned short* Bp = B + (size_t)nBase * ldb;

  FragF acc[2][4];
  for (int i = 0; i < 2; i++)
    for (int j = 0; j < 4; j++)
      for (int e = 0; e < 8; e++) acc[i][j].f[e] = 0.f;

  // prologue: tile 0 -> buffer 0
  {
    U4 ra0 = *(const U4*)&Ap[(size_t)r0 * lda + c0];
    U4 ra1 = *(const U4*)&Ap[(size_t)(r0 + 64) * lda + c0];
    U4 rb0 = *(const U4*)&Bp[(size_t)r0 * ldb + c0];
    U4 rb1 = *(const U4*)&Bp[(size_t)(r0 + 64) * ldb + c0];
    *(U4*)&lA[0][r0 * 40 + c0] = ra0; *(U4*)&lA[0][(r0 + 64) * 40 + c0] = ra1;
    *(U4*)&lB[0][r0 * 40 + c0] = rb0; *(U4*)&lB[0][(r0 + 64) * 40 + c0] = rb1;
  }
  __syncthreads();

  int buf = 0;
  for (int k0 = 0; k0 < K; k0 += 32) {
    const int kn = k0 + 32;
    const bool more = kn < K;
    U4 na0, na1, nb0, nb1;
    if (more) {                                           // prefetch next tile into regs
      na0 = *(const U4*)&Ap[(size_t)r0 * lda + kn + c0];
      na1 = *(const U4*)&Ap[(size_t)(r0 + 64) * lda + kn + c0];
      nb0 = *(const U4*)&Bp[(size_t)r0 * ldb + kn + c0];
      nb1 = *(const U4*)&Bp[(size_t)(r0 + 64) * ldb + kn + c0];
      if (kn + 32 < K) {                                  // L2 prefetch tile after next
        __builtin_prefetch(&Ap[(size_t)r0 * lda + kn + 32 + c0], 0, 1);
        __builtin_prefetch(&Bp[(size_t)r0 * ldb + kn + 32 + c0], 0, 1);
      }
    }
    const unsigned short* la = lA[buf];
    const unsigned short* lb = lB[buf];
    FragBF a[2], b[4];
    for (int i = 0; i < 2; i++) {                         // A frag: K chunks {laneH*8, +16}
      int r = wm + i * 16 + laneN, cb = laneH * 8;
      a[i].q[0] = *(const U4*)&la[r * 40 + cb];
      a[i].q[1] = *(const U4*)&la[r * 40 + cb + 16];
    }
    for (int j = 0; j < 4; j++) {                         // B frag: K chunks {laneH*8, +16}
      int r = wn + j * 16 + laneN, cb = laneH * 8;
      b[j].q[0] = *(const U4*)&lb[r * 40 + cb];
      b[j].q[1] = *(const U4*)&lb[r * 40 + cb + 16];
    }
    for (int i = 0; i < 2; i++)
      for (int j = 0; j < 4; j++)
        acc[i][j].v = __builtin_amdgcn_wmma_f32_16x16x32_bf16(
            false, a[i].v, false, b[j].v, (short)0, acc[i][j].v, false, false);
    if (more) {                                           // stage regs -> other buffer
      *(U4*)&lA[buf ^ 1][r0 * 40 + c0] = na0; *(U4*)&lA[buf ^ 1][(r0 + 64) * 40 + c0] = na1;
      *(U4*)&lB[buf ^ 1][r0 * 40 + c0] = nb0; *(U4*)&lB[buf ^ 1][(r0 + 64) * 40 + c0] = nb1;
    }
    __syncthreads();
    buf ^= 1;
  }
  for (int i = 0; i < 2; i++)
    for (int j = 0; j < 4; j++) {
      int row0 = mBase + wm + i * 16 + laneH * 8;
      int col = nBase + wn + j * 16 + laneN;
      for (int v = 0; v < 8; v++) {
        size_t off = (size_t)(row0 + v) * ldc + col;
        float val = acc[i][j].f[v];
        if (EPI == 0) Cf[off] = val;
        else if (EPI == 1) Cf[off] = val + resid[off];
        else { float r = val > 0.f ? val : 0.f; Cbf[off] = f2bf(r * r); }
      }
    }
}

// ---------------- flash attention (per-wave 16 query rows) -------------------------
__global__ __launch_bounds__(256) void flash_attn(const unsigned short* __restrict__ q,
                                                  const unsigned short* __restrict__ kmat,
                                                  const unsigned short* __restrict__ vt,
                                                  const int* __restrict__ docs,
                                                  const float* __restrict__ attn_scale,
                                                  const float* __restrict__ agate,
                                                  unsigned short* __restrict__ y) {
  __shared__ alignas(16) unsigned short pbuf[8 * 16 * 40];  // wave-private P relayout tiles
  const int tid = threadIdx.x, wid = tid >> 5, lane = tid & 31;
  const int laneN = lane & 15, laneH = lane >> 4;
  const int head = blockIdx.y;
  const int qBase = blockIdx.x * 128 + wid * 16;
  unsigned short* pb = &pbuf[wid * 16 * 40];
  const float sc = attn_scale[0];

  FragBF aq[4];                               // Q row tile 16x128 resident
  {
    int row = qBase + laneN;
    const unsigned short* qp = q + ((size_t)row * NH + head) * HD;
    for (int c = 0; c < 4; c++) {
      int d0 = c * 32 + laneH * 8;
      aq[c].q[0] = *(const U4*)&qp[d0];
      aq[c].q[1] = *(const U4*)&qp[d0 + 16];
    }
  }
  int tq[8], dq[8]; float gate[8];
  for (int v = 0; v < 8; v++) {
    tq[v] = qBase + v + 8 * laneH;            // C-layout row mapping
    dq[v] = docs[tq[v]];
    gate[v] = agate[tq[v] * NH + head];
  }
  FragF O[8];
  for (int j = 0; j < 8; j++) for (int e = 0; e < 8; e++) O[j].f[e] = 0.f;
  float m[8], l[8];
  for (int v = 0; v < 8; v++) { m[v] = -1e30f; l[v] = 0.f; }

  const int kmax = blockIdx.x * 128 + 128;    // causal: uniform bound over the block
  for (int s0 = 0; s0 < kmax; s0 += 32) {
    FragF SA, SB;
    for (int e = 0; e < 8; e++) { SA.f[e] = 0.f; SB.f[e] = 0.f; }
    FragBF bk[8];                             // batch all K fragments first
    {
      const unsigned short* kp  = kmat + ((size_t)(s0 + laneN) * NH + head) * HD;
      const unsigned short* kp2 = kmat + ((size_t)(s0 + 16 + laneN) * NH + head) * HD;
      for (int c = 0; c < 4; c++) {
        int d0 = c * 32 + laneH * 8;
        bk[c].q[0]     = *(const U4*)&kp[d0];  bk[c].q[1]     = *(const U4*)&kp[d0 + 16];
        bk[4 + c].q[0] = *(const U4*)&kp2[d0]; bk[4 + c].q[1] = *(const U4*)&kp2[d0 + 16];
      }
      if (s0 + 32 < kmax)
        __builtin_prefetch(kmat + ((size_t)(s0 + 32 + laneN) * NH + head) * HD, 0, 1);
    }
    for (int c = 0; c < 4; c++)
      SA.v = __builtin_amdgcn_wmma_f32_16x16x32_bf16(false, aq[c].v, false, bk[c].v, (short)0, SA.v, false, false);
    for (int c = 0; c < 4; c++)
      SB.v = __builtin_amdgcn_wmma_f32_16x16x32_bf16(false, aq[c].v, false, bk[4 + c].v, (short)0, SB.v, false, false);

    int sA = s0 + laneN, sB = sA + 16;
    int dA = docs[sA], dB = docs[sB];
    float alpha[8];
    for (int v = 0; v < 8; v++) {             // online softmax, half-wave xor reductions
      float a = (tq[v] >= sA && dq[v] == dA) ? SA.f[v] * sc : -1e30f;
      float b = (tq[v] >= sB && dq[v] == dB) ? SB.f[v] * sc : -1e30f;
      float rm = fmaxf(a, b);
      for (int msk = 1; msk < 16; msk <<= 1) rm = fmaxf(rm, __shfl_xor(rm, msk, 32));
      float mn = fmaxf(m[v], rm);
      alpha[v] = __expf(m[v] - mn);
      m[v] = mn;
      float pa = __expf(a - mn), pbv = __expf(b - mn);
      SA.f[v] = pa; SB.f[v] = pbv;
      float rs = pa + pbv;
      for (int msk = 1; msk < 16; msk <<= 1) rs += __shfl_xor(rs, msk, 32);
      l[v] = l[v] * alpha[v] + rs;
    }
    for (int j = 0; j < 8; j++)
      for (int v = 0; v < 8; v++) O[j].f[v] *= alpha[v];
    for (int v = 0; v < 8; v++) {             // P: C-layout -> LDS (bf16, 16x32)
      int row = v + 8 * laneH;
      pb[row * 40 + laneN] = f2bf(SA.f[v]);
      pb[row * 40 + 16 + laneN] = f2bf(SB.f[v]);
    }
    __syncthreads();
    FragBF ap;                                // read back in A-fragment layout
    {
      int cb = laneH * 8;
      ap.q[0] = *(const U4*)&pb[laneN * 40 + cb];
      ap.q[1] = *(const U4*)&pb[laneN * 40 + cb + 16];
    }
    for (int jj = 0; jj < 2; jj++) {          // O += P · V, V pre-transposed (H,HD,S)
      FragBF bv[4];
      for (int j = 0; j < 4; j++) {
        const unsigned short* vp =
            vt + (size_t)(head * HD + (jj * 4 + j) * 16 + laneN) * T_SEQ + s0 + laneH * 8;
        bv[j].q[0] = *(const U4*)&vp[0];
        bv[j].q[1] = *(const U4*)&vp[16];
      }
      for (int j = 0; j < 4; j++)
        O[jj * 4 + j].v = __builtin_amdgcn_wmma_f32_16x16x32_bf16(
            false, ap.v, false, bv[j].v, (short)0, O[jj * 4 + j].v, false, false);
    }
    __syncthreads();
  }
  for (int j = 0; j < 8; j++)
    for (int v = 0; v < 8; v++) {
      float val = O[j].f[v] / l[v] * gate[v];
      y[(size_t)tq[v] * DIM + head * HD + j * 16 + laneN] = f2bf(val);
    }
}

// ---------------- launch -----------------------------------------------------------
extern "C" void kernel_launch(void* const* d_in, const int* in_sizes, int n_in,
                              void* d_out, int out_size, void* d_ws, size_t ws_size,
                              hipStream_t stream) {
  (void)in_sizes; (void)n_in; (void)out_size; (void)ws_size;
  const float* x       = (const float*)d_in[0];
  const float* ve      = (const float*)d_in[1];
  const float* qkvo_w  = (const float*)d_in[2];
  const float* agw     = (const float*)d_in[3];
  const float* vgw     = (const float*)d_in[4];
  const float* c_fc    = (const float*)d_in[5];
  const float* c_proj  = (const float*)d_in[6];
  const float* lambdas = (const float*)d_in[7];
  const float* cosb    = (const float*)d_in[8];
  const float* sinb    = (const float*)d_in[9];
  const float* ascale  = (const float*)d_in[10];
  const int*   docs    = (const int*)d_in[11];
  const int*   koff    = (const int*)d_in[12];
  float* out = (float*)d_out;

  char* ws = (char*)d_ws;
  size_t off = 0;
  auto alloc = [&](size_t bytes) -> void* {
    void* p = ws + off;
    off += (bytes + 255) & ~(size_t)255;
    return p;
  };
  unsigned short* xn   = (unsigned short*)alloc((size_t)T_SEQ * DIM * 2);
  unsigned short* wqkv = (unsigned short*)alloc((size_t)E3 * DIM * 2);
  unsigned short* wo   = (unsigned short*)alloc((size_t)DIM * DIM * 2);
  unsigned short* wfc  = (unsigned short*)alloc((size_t)FFN * DIM * 2);
  unsigned short* wpt  = (unsigned short*)alloc((size_t)DIM * FFN * 2);
  float*          qkvf = (float*)alloc((size_t)T_SEQ * E3 * 4);
  float*          agate= (float*)alloc((size_t)T_SEQ * NH * 4);
  float*          vgate= (float*)alloc((size_t)T_SEQ * NH * 4);
  unsigned short* qb   = (unsigned short*)alloc((size_t)T_SEQ * DIM * 2);
  unsigned short* kb   = (unsigned short*)alloc((size_t)T_SEQ * DIM * 2);
  unsigned short* vtb  = (unsigned short*)alloc((size_t)T_SEQ * DIM * 2);
  unsigned short* yb   = (unsigned short*)alloc((size_t)T_SEQ * DIM * 2);
  float*          xmid = (float*)alloc((size_t)T_SEQ * DIM * 4);
  unsigned short* xn2  = (unsigned short*)alloc((size_t)T_SEQ * DIM * 2);
  unsigned short* hb   = (unsigned short*)alloc((size_t)T_SEQ * FFN * 2);

  // weight prep (bf16, lambda-scaled, c_proj transposed)
  convert_scale_kernel<<<((size_t)E3 * DIM + 255) / 256, 256, 0, stream>>>(qkvo_w, wqkv, E3 * DIM, lambdas, 0);
  convert_scale_kernel<<<((size_t)DIM * DIM + 255) / 256, 256, 0, stream>>>(qkvo_w + (size_t)E3 * DIM, wo, DIM * DIM, lambdas, 1);
  convert_scale_kernel<<<((size_t)FFN * DIM + 255) / 256, 256, 0, stream>>>(c_fc, wfc, FFN * DIM, nullptr, 0);
  transpose_convert_kernel<<<dim3(FFN / 32, DIM / 32), 256, 0, stream>>>(c_proj, wpt);

  rmsnorm_kernel<<<T_SEQ, 256, 0, stream>>>(x, xn);
  gates_kernel<<<T_SEQ, 256, 0, stream>>>(xn, agw, vgw, agate, vgate);

  // qkv = xn · w_qkv^T   (2048 x 6144 x 2048)
  gemm_nt<0><<<dim3(E3 / 128, T_SEQ / 128), 256, 0, stream>>>(xn, DIM, wqkv, DIM, qkvf, nullptr, E3, nullptr, DIM);

  qk_prep<<<dim3(T_SEQ, NH), 128, 0, stream>>>(qkvf, cosb, sinb, koff, qb, kb);
  v_prep<<<dim3(T_SEQ / 64, NH), 256, 0, stream>>>(qkvf, ve, vgate, vtb);

  flash_attn<<<dim3(T_SEQ / 128, NH), 256, 0, stream>>>(qb, kb, vtb, docs, ascale, agate, yb);

  // x_mid = x + y · w_o^T
  gemm_nt<1><<<dim3(DIM / 128, T_SEQ / 128), 256, 0, stream>>>(yb, DIM, wo, DIM, xmid, nullptr, DIM, x, DIM);

  rmsnorm_kernel<<<T_SEQ, 256, 0, stream>>>(xmid, xn2);

  // h = relu(xn2 · c_fc^T)^2  -> bf16
  gemm_nt<2><<<dim3(FFN / 128, T_SEQ / 128), 256, 0, stream>>>(xn2, DIM, wfc, DIM, nullptr, hb, FFN, nullptr, DIM);

  // out = x_mid + h · c_proj  (via transposed weight, K=8192)
  gemm_nt<1><<<dim3(DIM / 128, T_SEQ / 128), 256, 0, stream>>>(hb, FFN, wpt, FFN, out, nullptr, DIM, xmid, FFN);
}
